// AttentionBlock_40759239639710
// MI455X (gfx1250) — matrix operand order repaired and checked by reading, not compiled, not run on hardware
//
#include <hip/hip_runtime.h>

// ---------------- types ----------------
typedef unsigned short ushort_t;
typedef __attribute__((ext_vector_type(16))) __bf16        bf16x16;
typedef __attribute__((ext_vector_type(8)))  float         f32x8;
typedef __attribute__((ext_vector_type(4)))  unsigned int  u32x4;

union FragB16 { bf16x16 v; u32x4 q[2]; ushort_t s[16]; };

__device__ __forceinline__ ushort_t f2bf(float f) {
    unsigned u = __float_as_uint(f);
    return (ushort_t)((u + 0x7FFFu + ((u >> 16) & 1u)) >> 16);   // RNE
}

// CDNA5 async copy: global -> LDS, 16 bytes per lane, tracked by ASYNCcnt.
__device__ __forceinline__ void async_ld_b128(unsigned lds_off, const void* g) {
    asm volatile("global_load_async_to_lds_b128 %0, %1, off"
                 :: "v"(lds_off), "v"((unsigned long long)(size_t)g)
                 : "memory");
}
__device__ __forceinline__ void wait_async0() {
    asm volatile("s_wait_asynccnt 0x0" ::: "memory");
}

// ---------------- problem constants ----------------
#define Bz   8
#define Cc   256
#define Nn   4096          // 64*64
#define Gg   8
#define CPG  32            // channels per group
#define EPSI 1e-5f

// workspace byte offsets
#define OFF_STATS 0
#define OFF_WQB   (4096)
#define OFF_WKB   (OFF_WQB + 131072)
#define OFF_WVB   (OFF_WKB + 131072)
#define OFF_WPB   (OFF_WVB + 131072)
#define OFF_XNT   (1048576)                      // [B][N][C] bf16, 16 MB
#define OFF_QT    (OFF_XNT + 16777216)
#define OFF_KT    (OFF_QT  + 16777216)
#define OFF_V2    (OFF_KT  + 16777216)           // V in [B][C][N] bf16
#define OFF_OT    (OFF_V2  + 16777216)

// ---------------- 1) group-norm statistics ----------------
__global__ void gn_stats_k(const float* __restrict__ x, float* __restrict__ stats) {
    __shared__ float s_sum[256], s_sq[256];
    int bg = blockIdx.x;                                   // b*G + g
    const float* p = x + (size_t)bg * CPG * Nn;            // group is contiguous in [C][N]
    float sum = 0.f, sq = 0.f;
    for (int i = threadIdx.x; i < CPG * Nn; i += 256) {
        float v = p[i]; sum += v; sq += v * v;
    }
    s_sum[threadIdx.x] = sum; s_sq[threadIdx.x] = sq;
    __syncthreads();
    for (int st = 128; st > 0; st >>= 1) {
        if ((int)threadIdx.x < st) {
            s_sum[threadIdx.x] += s_sum[threadIdx.x + st];
            s_sq [threadIdx.x] += s_sq [threadIdx.x + st];
        }
        __syncthreads();
    }
    if (threadIdx.x == 0) {
        float inv = 1.0f / (float)(CPG * Nn);
        float m   = s_sum[0] * inv;
        float var = s_sq[0] * inv - m * m;
        stats[bg * 2 + 0] = m;
        stats[bg * 2 + 1] = rsqrtf(var + EPSI);
    }
}

// ---------------- 2) apply GN, write transposed bf16 xnT[b][n][c] ----------------
__global__ void gn_apply_k(const float* __restrict__ x, const float* __restrict__ gw,
                           const float* __restrict__ gb, const float* __restrict__ stats,
                           ushort_t* __restrict__ xnT) {
    int idx = blockIdx.x * 256 + threadIdx.x;   // b*(N*C) + n*C + c
    int c = idx & (Cc - 1);
    int n = (idx >> 8) & (Nn - 1);
    int b = idx >> 20;
    int g = c >> 5;
    float m = stats[(b * Gg + g) * 2 + 0];
    float r = stats[(b * Gg + g) * 2 + 1];
    float v = x[((size_t)(b * Cc + c) << 12) + n];
    xnT[idx] = f2bf((v - m) * r * gw[c] + gb[c]);
}

// ---------------- 3) weights -> bf16 ----------------
__global__ void wprep_k(const float* __restrict__ wq, const float* __restrict__ wk,
                        const float* __restrict__ wv, const float* __restrict__ wp,
                        ushort_t* __restrict__ oq, ushort_t* __restrict__ ok,
                        ushort_t* __restrict__ ov, ushort_t* __restrict__ op) {
    int i = blockIdx.x * 256 + threadIdx.x;     // 0..65535
    oq[i] = f2bf(wq[i]); ok[i] = f2bf(wk[i]); ov[i] = f2bf(wv[i]); op[i] = f2bf(wp[i]);
}

// ---------------- 4) q/k/v projections ----------------
// q,k stored [b][n][c]; v stored [b][c][n] so the PV B-fragment is j-contiguous.
__global__ void __launch_bounds__(128)
qkv_k(const ushort_t* __restrict__ xnT,
      const ushort_t* __restrict__ wqb, const ushort_t* __restrict__ wkb,
      const ushort_t* __restrict__ wvb,
      const float* __restrict__ bq, const float* __restrict__ bk, const float* __restrict__ bv,
      ushort_t* __restrict__ qT, ushort_t* __restrict__ kT, ushort_t* __restrict__ v2) {
    int lane = threadIdx.x & 31, wid = blockIdx.x * 4 + (threadIdx.x >> 5);
    int mat = wid >> 15;            // 32768 tiles per matrix
    int rem = wid & 32767;
    int b   = rem >> 12;            // 4096 tiles per batch
    int t   = rem & 4095;
    int nt  = t >> 4, ot = t & 15;
    const ushort_t* w  = (mat == 0) ? wqb : (mat == 1) ? wkb : wvb;
    const float*    bb = (mat == 0) ? bq  : (mat == 1) ? bk  : bv;
    float scl = (mat == 0) ? 0.0625f : 1.0f;    // fold 1/sqrt(C) into q
    int hi = lane >> 4, lo = lane & 15;

    f32x8 acc = {};
    const ushort_t* arow = xnT + ((size_t)(b * Nn + nt * 16 + lo)) * Cc + hi * 8;
    const ushort_t* brow = w + (ot * 16 + lo) * Cc + hi * 8;
#pragma unroll
    for (int ks = 0; ks < 8; ++ks) {
        FragB16 a, bfr;
        a.q[0]   = *(const u32x4*)(arow + ks * 32);
        a.q[1]   = *(const u32x4*)(arow + ks * 32 + 16);
        bfr.q[0] = *(const u32x4*)(brow + ks * 32);
        bfr.q[1] = *(const u32x4*)(brow + ks * 32 + 16);
        acc = __builtin_amdgcn_wmma_f32_16x16x32_bf16(false, a.v, false, bfr.v,
                                                      (short)0, acc, false, false);
    }
    int o = ot * 16 + lo;
    float bias = bb[o];
    if (mat == 2) {
        // v2[b][o][n]: 8 consecutive n per lane -> one 16B store
        union { u32x4 q; ushort_t s[8]; } st;
#pragma unroll
        for (int r = 0; r < 8; ++r) st.s[r] = f2bf(acc[r] + bias);
        *(u32x4*)(v2 + ((size_t)(b * Cc + o)) * Nn + nt * 16 + hi * 8) = st.q;
    } else {
        ushort_t* d = (mat == 0) ? qT : kT;
#pragma unroll
        for (int r = 0; r < 8; ++r) {
            int n = nt * 16 + r + hi * 8;
            d[((size_t)(b * Nn + n)) * Cc + o] = f2bf((acc[r] + bias) * scl);
        }
    }
}

// ---------------- 5) flash attention: outT[n][c] = softmax(qT kT^T) V ----------------
__global__ void __launch_bounds__(128)
attn_k(const ushort_t* __restrict__ qT, const ushort_t* __restrict__ kT,
       const ushort_t* __restrict__ v2, ushort_t* __restrict__ outT) {
    __shared__ u32x4 ldsK4[32 * 32];          // K tile [32 j][256 c] bf16, 16 KB
    __shared__ u32x4 ldsV4[256 * 4];          // V tile [256 c][32 j] bf16, 16 KB
    __shared__ float sS[4][16 * 32];          // per-wave raw scores
    __shared__ u32x4 sP4[4][64];              // per-wave P (bf16, 16x32)
    __shared__ float sM[4][16], sL[4][16], sA[4][16];

    ushort_t* ldsK = (ushort_t*)ldsK4;
    ushort_t* ldsV = (ushort_t*)ldsV4;

    int tid = threadIdx.x;
    int lane = tid & 31, w = tid >> 5;
    int hi = lane >> 4, lo = lane & 15;
    int b = blockIdx.x >> 6;
    int i0 = ((blockIdx.x & 63) * 4 + w) * 16;

    const ushort_t* qb  = qT + (size_t)b * Nn * Cc;
    const ushort_t* kb  = kT + (size_t)b * Nn * Cc;
    const ushort_t* vb2 = v2 + (size_t)b * Cc * Nn;

    bf16x16 qA[8];
    {
        const ushort_t* arow = qb + (size_t)(i0 + lo) * Cc + hi * 8;
#pragma unroll
        for (int ks = 0; ks < 8; ++ks) {
            FragB16 a;
            a.q[0] = *(const u32x4*)(arow + ks * 32);
            a.q[1] = *(const u32x4*)(arow + ks * 32 + 16);
            qA[ks] = a.v;
        }
    }
    f32x8 zero = {};
    f32x8 acc[16];
#pragma unroll
    for (int ct = 0; ct < 16; ++ct) acc[ct] = zero;
    if (lane < 16) { sM[w][lane] = -3.0e38f; sL[w][lane] = 0.f; }

    for (int js = 0; js < 128; ++js) {
        int j0 = js * 32;
        __syncthreads();   // previous iteration's LDS reads complete
        // async-stage K tile [32 j][256 c] and V tile [256 c][32 j]
        for (int idx = tid; idx < 1024; idx += 128) {
            int jl = idx >> 5, c8 = (idx & 31) * 8;
            async_ld_b128((unsigned)(size_t)&ldsK4[idx],
                          kb + (size_t)(j0 + jl) * Cc + c8);
            int cr = idx >> 2, part = idx & 3;
            async_ld_b128((unsigned)(size_t)&ldsV4[idx],
                          vb2 + (size_t)cr * Nn + j0 + part * 8);
        }
        wait_async0();
        __syncthreads();

        // scores: two 16x16 tiles over K=256
#pragma unroll
        for (int jt = 0; jt < 2; ++jt) {
            f32x8 s = zero;
            const ushort_t* brow = ldsK + (jt * 16 + lo) * 256 + hi * 8;
#pragma unroll
            for (int ks = 0; ks < 8; ++ks) {
                FragB16 bf;
                bf.q[0] = *(const u32x4*)(brow + ks * 32);
                bf.q[1] = *(const u32x4*)(brow + ks * 32 + 16);
                s = __builtin_amdgcn_wmma_f32_16x16x32_bf16(false, qA[ks], false, bf.v,
                                                            (short)0, s, false, false);
            }
#pragma unroll
            for (int r = 0; r < 8; ++r)
                sS[w][(r + hi * 8) * 32 + jt * 16 + lo] = s[r];
        }

        // online softmax update, one row per lane (lanes 0..15)
        if (lane < 16) {
            int m = lane;
            float mold = sM[w][m], l = sL[w][m];
            float sv[32]; float rmax = -3.0e38f;
#pragma unroll
            for (int j = 0; j < 32; ++j) { sv[j] = sS[w][m * 32 + j]; rmax = fmaxf(rmax, sv[j]); }
            float mnew = fmaxf(mold, rmax);
            float alpha = __expf(mold - mnew);
            float psum = 0.f;
            union { u32x4 q[4]; ushort_t s[32]; } pb;
#pragma unroll
            for (int j = 0; j < 32; ++j) { float p = __expf(sv[j] - mnew); psum += p; pb.s[j] = f2bf(p); }
            sM[w][m] = mnew; sL[w][m] = l * alpha + psum; sA[w][m] = alpha;
#pragma unroll
            for (int qi = 0; qi < 4; ++qi) sP4[w][m * 4 + qi] = pb.q[qi];
        }
        // wave-internal LDS ordering: no barrier needed before reading sA/sP4

        float al[8];
#pragma unroll
        for (int r = 0; r < 8; ++r) al[r] = sA[w][r + hi * 8];
#pragma unroll
        for (int ct = 0; ct < 16; ++ct)
#pragma unroll
            for (int r = 0; r < 8; ++r) acc[ct][r] *= al[r];

        FragB16 pA;
        {
            const ushort_t* prow = (const ushort_t*)sP4[w] + lo * 32 + hi * 8;
            pA.q[0] = *(const u32x4*)(prow);
            pA.q[1] = *(const u32x4*)(prow + 16);
        }
#pragma unroll
        for (int ct = 0; ct < 16; ++ct) {
            FragB16 bf;
            const ushort_t* vrow = ldsV + (ct * 16 + lo) * 32 + hi * 8;
            bf.q[0] = *(const u32x4*)(vrow);
            bf.q[1] = *(const u32x4*)(vrow + 16);
            acc[ct] = __builtin_amdgcn_wmma_f32_16x16x32_bf16(false, pA.v, false, bf.v,
                                                              (short)0, acc[ct], false, false);
        }
    }

    // normalize and store outT[n][c] bf16
    float invl[8];
#pragma unroll
    for (int r = 0; r < 8; ++r) invl[r] = 1.0f / sL[w][r + hi * 8];
    ushort_t* ob = outT + (size_t)b * Nn * Cc;
#pragma unroll
    for (int ct = 0; ct < 16; ++ct) {
        int c = ct * 16 + lo;
#pragma unroll
        for (int r = 0; r < 8; ++r) {
            int n = i0 + r + hi * 8;
            ob[(size_t)n * Cc + c] = f2bf(acc[ct][r] * invl[r]);
        }
    }
}

// ---------------- 6) output projection + residual ----------------
__global__ void __launch_bounds__(128)
proj_k(const ushort_t* __restrict__ outT, const ushort_t* __restrict__ wpb,
       const float* __restrict__ bp, const float* __restrict__ x,
       float* __restrict__ y) {
    int lane = threadIdx.x & 31, wid = blockIdx.x * 4 + (threadIdx.x >> 5);
    int b = wid >> 12;
    int t = wid & 4095;
    int nt = t >> 4, ot = t & 15;
    int hi = lane >> 4, lo = lane & 15;
    f32x8 acc = {};
    const ushort_t* arow = outT + ((size_t)(b * Nn + nt * 16 + lo)) * Cc + hi * 8;
    const ushort_t* brow = wpb + (ot * 16 + lo) * Cc + hi * 8;
#pragma unroll
    for (int ks = 0; ks < 8; ++ks) {
        FragB16 a, bf;
        a.q[0]  = *(const u32x4*)(arow + ks * 32);
        a.q[1]  = *(const u32x4*)(arow + ks * 32 + 16);
        bf.q[0] = *(const u32x4*)(brow + ks * 32);
        bf.q[1] = *(const u32x4*)(brow + ks * 32 + 16);
        acc = __builtin_amdgcn_wmma_f32_16x16x32_bf16(false, a.v, false, bf.v,
                                                      (short)0, acc, false, false);
    }
    int o = ot * 16 + lo;
    float bias = bp[o];
    size_t base = (size_t)b * Cc * Nn + (size_t)o * Nn + nt * 16;
#pragma unroll
    for (int r = 0; r < 8; ++r) {
        int n = r + hi * 8;
        y[base + n] = x[base + n] + acc[r] + bias;
    }
}

// ---------------- host launcher ----------------
extern "C" void kernel_launch(void* const* d_in, const int* in_sizes, int n_in,
                              void* d_out, int out_size, void* d_ws, size_t ws_size,
                              hipStream_t stream) {
    const float* x   = (const float*)d_in[0];
    const float* gnw = (const float*)d_in[1];
    const float* gnb = (const float*)d_in[2];
    const float* wq  = (const float*)d_in[3];
    const float* bq  = (const float*)d_in[4];
    const float* wk  = (const float*)d_in[5];
    const float* bk  = (const float*)d_in[6];
    const float* wv  = (const float*)d_in[7];
    const float* bv  = (const float*)d_in[8];
    const float* wp  = (const float*)d_in[9];
    const float* bp  = (const float*)d_in[10];
    float* out = (float*)d_out;

    char* ws = (char*)d_ws;
    float*    stats = (float*)   (ws + OFF_STATS);
    ushort_t* wqb   = (ushort_t*)(ws + OFF_WQB);
    ushort_t* wkb   = (ushort_t*)(ws + OFF_WKB);
    ushort_t* wvb   = (ushort_t*)(ws + OFF_WVB);
    ushort_t* wpb   = (ushort_t*)(ws + OFF_WPB);
    ushort_t* xnT   = (ushort_t*)(ws + OFF_XNT);
    ushort_t* qT    = (ushort_t*)(ws + OFF_QT);
    ushort_t* kT    = (ushort_t*)(ws + OFF_KT);
    ushort_t* v2    = (ushort_t*)(ws + OFF_V2);
    ushort_t* oT    = (ushort_t*)(ws + OFF_OT);

    gn_stats_k<<<dim3(Bz * Gg), dim3(256), 0, stream>>>(x, stats);
    gn_apply_k<<<dim3((Bz * Nn * Cc) / 256), dim3(256), 0, stream>>>(x, gnw, gnb, stats, xnT);
    wprep_k<<<dim3(256), dim3(256), 0, stream>>>(wq, wk, wv, wp, wqb, wkb, wvb, wpb);
    qkv_k<<<dim3(24576), dim3(128), 0, stream>>>(xnT, wqb, wkb, wvb, bq, bk, bv, qT, kT, v2);
    attn_k<<<dim3(512), dim3(128), 0, stream>>>(qT, kT, v2, oT);
    proj_k<<<dim3(8192), dim3(128), 0, stream>>>(oT, wpb, bp, x, out);
}
